// Discriminator_13889924235980
// MI455X (gfx1250) — matrix operand order repaired and checked by reading, not compiled, not run on hardware
//
#include <hip/hip_runtime.h>
#include <cmath>

typedef __attribute__((ext_vector_type(16))) _Float16 v16h;
typedef __attribute__((ext_vector_type(8)))  _Float16 v8h;
typedef __attribute__((ext_vector_type(8)))  float    v8f;

#define WAVES_PER_BLOCK 8

// ---------------------------------------------------------------------------
// A-fragment load for V_WMMA_F32_16X16X32_F16 (16-bit A, 16x32):
// lanes 0-15 : row M=lane,    K[kk+0..15]
// lanes 16-31: row M=lane-16, K[kk+16..31]
// Arow is pre-offset by (hi ? 16 : 0), so two contiguous 16B loads suffice.
// ---------------------------------------------------------------------------
__device__ __forceinline__ v16h load_a_frag(const _Float16* __restrict__ Arow, int kk) {
    v8h lo = *(const v8h*)(Arow + kk);
    v8h hi = *(const v8h*)(Arow + kk + 8);
    v16h a;
#pragma unroll
    for (int i = 0; i < 8; ++i) { a[i] = lo[i]; a[i + 8] = hi[i]; }
    return a;
}

// ---------------------------------------------------------------------------
// Implicit-GEMM forward conv (k=4) + fused FISTA shrink epilogue.
//   out[m][n] = relu( (HASY ? yb[m][n] : 0) + ss * sum_k A[m][k]*patch[k][n] - thr )
// One wave computes (16*MT) x 16 of the output. Per K-chunk: MT A-fragments
// are preloaded as one clause, the NEXT chunk's B gather is issued, then MT
// independent WMMAs retire the current chunk (software pipeline).
// Gather is branchless: clamped always-valid addresses + mask select; the
// K tail / pipeline overshoot is handled by ic-clamp + (kbase < K) mask.
// ---------------------------------------------------------------------------
template <int MT, bool HASY, bool MGUARD>
__global__ __launch_bounds__(32 * WAVES_PER_BLOCK)
void conv_gemm_wmma(
    const _Float16* __restrict__ A, const float* __restrict__ act,
    const float* __restrict__ yb, float* __restrict__ out,
    int M, int N, int K, int Kpad,
    int IH, int IW, int OH, int OW,
    long long sC, long long sB, long long sH,
    int stride, int pad, float ss, float thr,
    int tiles_n, int total_tiles)
{
    int wave = threadIdx.x >> 5;
    int tile = blockIdx.x * WAVES_PER_BLOCK + wave;
    if (tile >= total_tiles) return;
    int m0 = (tile / tiles_n) * (16 * MT);
    int n0 = (tile % tiles_n) << 4;

    int lane = threadIdx.x & 31;
    int lrow = lane & 15;
    int hi   = (lane >> 4) & 1;

    // per-lane column (b, oh, ow) -> clamped offsets + masks for 4x4 patch
    int n   = n0 + lrow;
    int ohw = OH * OW;
    int b   = n / ohw;
    int r2  = n - b * ohw;
    int oh  = r2 / OW;
    int ow  = r2 - oh * OW;
    int ih0 = oh * stride - pad;
    int iw0 = ow * stride - pad;
    const float* actb = act + (long long)b * sB;

    int  soff[16];
    bool okm[16];
#pragma unroll
    for (int kh = 0; kh < 4; ++kh) {
#pragma unroll
        for (int kw = 0; kw < 4; ++kw) {
            int ih = ih0 + kh, iw = iw0 + kw;
            bool ok = ((unsigned)ih < (unsigned)IH) && ((unsigned)iw < (unsigned)IW);
            soff[kh * 4 + kw] = ok ? (ih * (int)sH + iw) : 0;   // clamped: always valid
            okm[kh * 4 + kw]  = ok;
        }
    }

    const _Float16* Abase = A + (long long)(m0 + lrow) * Kpad + (hi ? 16 : 0);
    const int kblo  = hi ? 16 : 0;
    const int maxic = (K >> 4) - 1;

    auto gather = [&](int kk) -> v16h {
        int  kbase = kk + kblo;
        int  icc   = kbase >> 4;
        bool kok   = kbase < K;                 // masks K tail & pipeline overshoot
        if (icc > maxic) icc = maxic;           // keep loads in-bounds
        const float* p = actb + (long long)icc * sC;
        v16h bf;
#pragma unroll
        for (int j = 0; j < 16; ++j) {
            float v = p[soff[j]];               // unconditional load
            bf[j] = (_Float16)((okm[j] && kok) ? v : 0.0f);
        }
        return bf;
    };

    v8f acc[MT];
#pragma unroll
    for (int mi = 0; mi < MT; ++mi) { v8f z = {}; acc[mi] = z; }

    v16h bf = gather(0);
    for (int kk = 0; kk < Kpad; kk += 32) {
        v16h a[MT];
#pragma unroll
        for (int mi = 0; mi < MT; ++mi)
            a[mi] = load_a_frag(Abase + (long long)mi * 16 * Kpad, kk);
        v16h bfn = gather(kk + 32);             // prefetch next chunk's B
#pragma unroll
        for (int mi = 0; mi < MT; ++mi)
            acc[mi] = __builtin_amdgcn_wmma_f32_16x16x32_f16(
                false, a[mi], false, bf, (short)0, acc[mi], false, false);
        bf = bfn;
    }

    // D layout: VGPR j -> row (tile_m + j + 8*hi), col n
#pragma unroll
    for (int mi = 0; mi < MT; ++mi) {
#pragma unroll
        for (int j = 0; j < 8; ++j) {
            int m = m0 + mi * 16 + j + (hi ? 8 : 0);
            if (!MGUARD || (m < M)) {
                size_t idx = (size_t)m * (size_t)N + (size_t)n;
                float v = ss * acc[mi][j] - thr + (HASY ? yb[idx] : 0.0f);
                out[idx] = fmaxf(v, 0.0f);
            }
        }
    }
}

// ---------------------------------------------------------------------------
// Implicit-GEMM transposed conv (k=4) + fused residual epilogue.
//   r[m][n] = xin[m][n] - sum_k Wt[m][k] * yg[k][n]
// Wt: f16 [ICpad][Kt], Wt[ic][oc*16+kh*4+kw] = wn[oc][ic][kh][kw]
// y:  f32 [OC][B][OH][OW] (oc stride = yCs); columns n = (b,ih,iw)
// Same structure / pipeline as conv_gemm_wmma; gather offsets are K-invariant.
// ---------------------------------------------------------------------------
template <int MT, bool MGUARD>
__global__ __launch_bounds__(32 * WAVES_PER_BLOCK)
void convt_gemm_wmma(
    const _Float16* __restrict__ A, const float* __restrict__ y,
    const float* __restrict__ xin, float* __restrict__ r,
    int M, int N, int Kt,
    int IH, int IW, int OH, int OW, int stride, int pad,
    long long yCs, long long xC, long long xB, long long xH,
    int tiles_n, int total_tiles)
{
    int wave = threadIdx.x >> 5;
    int tile = blockIdx.x * WAVES_PER_BLOCK + wave;
    if (tile >= total_tiles) return;
    int m0 = (tile / tiles_n) * (16 * MT);
    int n0 = (tile % tiles_n) << 4;

    int lane = threadIdx.x & 31;
    int lrow = lane & 15;
    int hi   = (lane >> 4) & 1;

    int n    = n0 + lrow;
    int ihiw = IH * IW;
    int b    = n / ihiw;
    int r2   = n - b * ihiw;
    int ih   = r2 / IW;
    int iw   = r2 - ih * IW;

    int  offs[16];
    bool okm[16];
#pragma unroll
    for (int kh = 0; kh < 4; ++kh) {
        int nh = ih + pad - kh;
        int oh2 = nh / stride;
        bool okh = (nh >= 0) && (nh % stride == 0) && (oh2 < OH);
#pragma unroll
        for (int kw = 0; kw < 4; ++kw) {
            int nw = iw + pad - kw;
            int ow2 = nw / stride;
            bool ok = okh && (nw >= 0) && (nw % stride == 0) && (ow2 < OW);
            offs[kh * 4 + kw] = ok ? (oh2 * OW + ow2) : 0;   // clamped: always valid
            okm[kh * 4 + kw]  = ok;
        }
    }
    const float* ybase = y + (long long)b * (OH * OW);

    const _Float16* Abase = A + (long long)(m0 + lrow) * Kt + (hi ? 16 : 0);
    const int kblo  = hi ? 16 : 0;
    const int maxoc = (Kt >> 4) - 1;

    auto gather = [&](int kk) -> v16h {
        int  kbase = kk + kblo;
        int  oc    = kbase >> 4;
        bool kok   = kbase < Kt;                // masks pipeline overshoot
        if (oc > maxoc) oc = maxoc;             // keep loads in-bounds
        const float* p = ybase + (long long)oc * yCs;
        v16h bf;
#pragma unroll
        for (int j = 0; j < 16; ++j) {
            float v = p[offs[j]];               // unconditional load
            bf[j] = (_Float16)((okm[j] && kok) ? v : 0.0f);
        }
        return bf;
    };

    v8f acc[MT];
#pragma unroll
    for (int mi = 0; mi < MT; ++mi) { v8f z = {}; acc[mi] = z; }

    v16h bf = gather(0);
    for (int kk = 0; kk < Kt; kk += 32) {
        v16h a[MT];
#pragma unroll
        for (int mi = 0; mi < MT; ++mi)
            a[mi] = load_a_frag(Abase + (long long)mi * 16 * Kt, kk);
        v16h bfn = gather(kk + 32);             // prefetch next chunk's B
#pragma unroll
        for (int mi = 0; mi < MT; ++mi)
            acc[mi] = __builtin_amdgcn_wmma_f32_16x16x32_f16(
                false, a[mi], false, bf, (short)0, acc[mi], false, false);
        bf = bfn;
    }

    long long xoff = (long long)b * xB + (long long)ih * xH + (long long)iw;
#pragma unroll
    for (int mi = 0; mi < MT; ++mi) {
#pragma unroll
        for (int j = 0; j < 8; ++j) {
            int m = m0 + mi * 16 + j + (hi ? 8 : 0);
            if (!MGUARD || (m < M)) {
                size_t idx = (size_t)m * (size_t)N + (size_t)n;
                r[idx] = xin[(long long)m * xC + xoff] - acc[mi][j];
            }
        }
    }
}

// ---------------------------------------------------------------------------
// Dictionary-atom L2 normalize -> f16 A-matrices for both GEMMs.
// wn[oc][ic*16 + kh*4 + kw], wt[ic][oc*16 + kh*4 + kw]; buffers pre-zeroed.
// ---------------------------------------------------------------------------
__global__ void prep_weights(const float* __restrict__ w,
                             _Float16* __restrict__ wn, _Float16* __restrict__ wt,
                             int IC, int Kpad, int Ktpad)
{
    int oc = blockIdx.x;
    int K = IC * 16;
    const float* wr = w + (size_t)oc * K;
    __shared__ float red[256];
    float s = 0.0f;
    for (int i = threadIdx.x; i < K; i += blockDim.x) { float v = wr[i]; s += v * v; }
    red[threadIdx.x] = s;
    __syncthreads();
    for (int st = 128; st > 0; st >>= 1) {
        if ((int)threadIdx.x < st) red[threadIdx.x] += red[threadIdx.x + st];
        __syncthreads();
    }
    float rn = rsqrtf(red[0]);
    for (int i = threadIdx.x; i < K; i += blockDim.x) {
        _Float16 hv = (_Float16)(wr[i] * rn);
        wn[(size_t)oc * Kpad + i] = hv;
        int ic = i >> 4, rr = i & 15;
        wt[(size_t)ic * Ktpad + oc * 16 + rr] = hv;
    }
}

// y = c_pre + coef * (c_pre - c_pre_pre)
__global__ void fista_y(float* __restrict__ y, const float* __restrict__ cp,
                        const float* __restrict__ cpp, float coef, int n)
{
    int i = blockIdx.x * blockDim.x + threadIdx.x;
    if (i < n) { float a = cp[i]; y[i] = a + coef * (a - cpp[i]); }
}

__global__ void lrelu_kernel(const float* __restrict__ in, float* __restrict__ out, int n)
{
    int i = blockIdx.x * blockDim.x + threadIdx.x;
    if (i < n) { float v = in[i]; out[i] = (v >= 0.0f) ? v : 0.2f * v; }
}

// Training-mode BN (batch stats, biased var) + leaky ReLU.
// Layout [C][B][H][W]: one block per channel, N contiguous elements.
__global__ void bn_lrelu(const float* __restrict__ in, float* __restrict__ out,
                         const float* __restrict__ g, const float* __restrict__ b, int N)
{
    int ch = blockIdx.x;
    const float* p = in + (size_t)ch * N;
    __shared__ float s1[256], s2[256];
    float a = 0.0f, q = 0.0f;
    for (int i = threadIdx.x; i < N; i += 256) { float v = p[i]; a += v; q += v * v; }
    s1[threadIdx.x] = a; s2[threadIdx.x] = q;
    __syncthreads();
    for (int st = 128; st > 0; st >>= 1) {
        if ((int)threadIdx.x < st) {
            s1[threadIdx.x] += s1[threadIdx.x + st];
            s2[threadIdx.x] += s2[threadIdx.x + st];
        }
        __syncthreads();
    }
    float mean = s1[0] / (float)N;
    float var  = s2[0] / (float)N - mean * mean;
    float rstd = rsqrtf(var + 1e-5f);
    float gg = g[ch], bb = b[ch];
    for (int i = threadIdx.x; i < N; i += 256) {
        float v = gg * (p[i] - mean) * rstd + bb;
        out[(size_t)ch * N + i] = (v >= 0.0f) ? v : 0.2f * v;
    }
}

// h: [256(nz)][256(b)] -> out[b][nz] = h[nz][b] / max(||h[:,b]||, 1e-12)
__global__ void l2norm_t(const float* __restrict__ h, float* __restrict__ out)
{
    int col = blockIdx.x;       // batch sample
    int t   = threadIdx.x;      // channel
    float v = h[(size_t)t * 256 + col];
    __shared__ float red[256];
    red[t] = v * v;
    __syncthreads();
    for (int st = 128; st > 0; st >>= 1) {
        if (t < st) red[t] += red[t + st];
        __syncthreads();
    }
    float nrm = sqrtf(red[0]);
    out[(size_t)col * 256 + t] = v / fmaxf(nrm, 1e-12f);
}

// ---------------------------------------------------------------------------
// One DictBlock layer (FISTA, N_STEPS=4). Final result lands in c0.
// All conv shapes here have OC % 64 == 0 -> MT=4, no M guard.
// conv_t has M = IC: layer 1 (IC=3) uses MT=1 + guard, others MT=4 unguarded.
// ---------------------------------------------------------------------------
static void run_dict_layer(hipStream_t stream,
    const float* actin, long long aC, long long aB, long long aH,
    const float* w, int OC, int IC, int IH, int IW, int stride, int pad,
    _Float16* wn, _Float16* wt,
    float* c0, float* c1, float* c2, float* ybuf, float* rbuf)
{
    const int Bn = 256;
    const int OH = (IH + 2 * pad - 4) / stride + 1;
    const int OW = (IW + 2 * pad - 4) / stride + 1;
    const int K = IC * 16, Kpad = (K + 31) & ~31, Kt = OC * 16;
    const int ICpad = (IC + 15) & ~15;
    const int N = Bn * OH * OW, Nimg = Bn * IH * IW;
    const float ss = 0.1f, thr = 0.1f * 0.1f;   // MU, LAMBDA*MU

    hipMemsetAsync(wn, 0, (size_t)OC * Kpad * sizeof(_Float16), stream);
    hipMemsetAsync(wt, 0, (size_t)ICpad * Kt * sizeof(_Float16), stream);
    prep_weights<<<OC, 256, 0, stream>>>(w, wn, wt, IC, Kpad, Kt);

    auto conv = [&](const float* src, long long sC, long long sB, long long sH,
                    const float* yb, float* dst) {
        int tn = N / 16;
        int total = tn * (OC / 64);             // MT = 4 (OC % 64 == 0 always)
        int blocks = (total + WAVES_PER_BLOCK - 1) / WAVES_PER_BLOCK;
        if (yb) {
            conv_gemm_wmma<4, true, false><<<blocks, 32 * WAVES_PER_BLOCK, 0, stream>>>(
                wn, src, yb, dst, OC, N, K, Kpad, IH, IW, OH, OW,
                sC, sB, sH, stride, pad, ss, thr, tn, total);
        } else {
            conv_gemm_wmma<4, false, false><<<blocks, 32 * WAVES_PER_BLOCK, 0, stream>>>(
                wn, src, yb, dst, OC, N, K, Kpad, IH, IW, OH, OW,
                sC, sB, sH, stride, pad, ss, thr, tn, total);
        }
    };

    auto convt = [&](const float* yptr) {
        int tn = Nimg / 16, tm = ICpad / 16;
        if ((tm & 3) == 0) {
            int total = tn * (tm >> 2);
            convt_gemm_wmma<4, false><<<(total + WAVES_PER_BLOCK - 1) / WAVES_PER_BLOCK,
                                        32 * WAVES_PER_BLOCK, 0, stream>>>(
                wt, yptr, actin, rbuf, IC, Nimg, Kt, IH, IW, OH, OW, stride, pad,
                (long long)N, aC, aB, aH, tn, total);
        } else {
            int total = tn * tm;
            convt_gemm_wmma<1, true><<<(total + WAVES_PER_BLOCK - 1) / WAVES_PER_BLOCK,
                                       32 * WAVES_PER_BLOCK, 0, stream>>>(
                wt, yptr, actin, rbuf, IC, Nimg, Kt, IH, IW, OH, OW, stride, pad,
                (long long)N, aC, aB, aH, tn, total);
        }
    };

    // t = 0
    conv(actin, aC, aB, aH, nullptr, c0);

    float* tgt[4] = {nullptr, c1, c2, c0};
    float* cpp = c0;
    float* cpre = c0;
    float tk = 1.0f;
    for (int t = 1; t <= 3; ++t) {
        float tkp = tk;
        tk = 0.5f * (1.0f + sqrtf(1.0f + 4.0f * tkp * tkp));
        float coef = (tkp - 1.0f) / tk;
        const float* yptr;
        if (t == 1) {
            yptr = cpre;                         // coef == 0 -> y = c_pre
        } else {
            int nel = OC * N;
            fista_y<<<(nel + 255) / 256, 256, 0, stream>>>(ybuf, cpre, cpp, coef, nel);
            yptr = ybuf;
        }
        convt(yptr);                             // r = x - convT(y)   (fused)
        // c = relu(y + ss*conv(r) - thr)  (fused); r layout [IC][B][IH][IW]
        conv(rbuf, (long long)Bn * IH * IW, (long long)IH * IW, (long long)IW,
             yptr, tgt[t]);
        cpp = cpre; cpre = tgt[t];
    }
}

// ---------------------------------------------------------------------------
extern "C" void kernel_launch(void* const* d_in, const int* in_sizes, int n_in,
                              void* d_out, int out_size, void* d_ws, size_t ws_size,
                              hipStream_t stream)
{
    (void)in_sizes; (void)n_in; (void)out_size; (void)ws_size;
    const float* x  = (const float*)d_in[0];
    const float* w1 = (const float*)d_in[1];
    const float* w2 = (const float*)d_in[2];
    const float* w3 = (const float*)d_in[3];
    const float* w4 = (const float*)d_in[4];
    const float* g2 = (const float*)d_in[5];
    const float* b2 = (const float*)d_in[6];
    const float* g3 = (const float*)d_in[7];
    const float* b3 = (const float*)d_in[8];
    float* out = (float*)d_out;

    char* ws = (char*)d_ws;
    size_t off = 0;
    auto carve = [&](size_t bytes) -> void* {
        void* p = ws + off;
        off += (bytes + 255) & ~(size_t)255;
        return p;
    };
    const size_t CMAX = 8388608;  // max OC*N over layers (layer1: 128*65536)
    float* c0 = (float*)carve(CMAX * 4);
    float* c1 = (float*)carve(CMAX * 4);
    float* c2 = (float*)carve(CMAX * 4);
    float* yb = (float*)carve(CMAX * 4);
    float* rb = (float*)carve(CMAX * 4);                  // max IC*Nimg (layer2)
    float* a1 = (float*)carve((size_t)128 * 65536 * 4);   // layer1 output
    float* a2 = (float*)carve((size_t)256 * 16384 * 4);   // layer2 output
    float* a3 = (float*)carve((size_t)512 * 4096 * 4);    // layer3 output
    _Float16* wn = (_Float16*)carve((size_t)4194304 * 2);
    _Float16* wt = (_Float16*)carve((size_t)4194304 * 2);

    // Layer 1: x [B][3][32][32] -> c [128][B][16][16]; lrelu
    run_dict_layer(stream, x, 1024, 3072, 32, w1, 128, 3, 32, 32, 2, 1,
                   wn, wt, c0, c1, c2, yb, rb);
    { int n = 128 * 65536; lrelu_kernel<<<(n + 255) / 256, 256, 0, stream>>>(c0, a1, n); }

    // Layer 2: [128][B][16][16] -> [256][B][8][8]; BN + lrelu
    run_dict_layer(stream, a1, 65536, 256, 16, w2, 256, 128, 16, 16, 2, 1,
                   wn, wt, c0, c1, c2, yb, rb);
    bn_lrelu<<<256, 256, 0, stream>>>(c0, a2, g2, b2, 16384);

    // Layer 3: [256][B][8][8] -> [512][B][4][4]; BN + lrelu
    run_dict_layer(stream, a2, 16384, 64, 8, w3, 512, 256, 8, 8, 2, 1,
                   wn, wt, c0, c1, c2, yb, rb);
    bn_lrelu<<<512, 256, 0, stream>>>(c0, a3, g3, b3, 4096);

    // Layer 4: [512][B][4][4] -> [256][B][1][1]; then per-sample L2 normalize
    run_dict_layer(stream, a3, 4096, 16, 4, w4, 256, 512, 4, 4, 1, 0,
                   wn, wt, c0, c1, c2, yb, rb);
    l2norm_t<<<256, 256, 0, stream>>>(c0, out);
}
